// Checker_56951266344970
// MI455X (gfx1250) — 8-bit matrix operand order repaired and checked by reading, compiled, not run on hardware
//
#include <hip/hip_runtime.h>
#include <hip/hip_bf16.h>

typedef __attribute__((ext_vector_type(8))) int v8i;
typedef __attribute__((ext_vector_type(4))) int v4i;

#define CIN   256
#define COUT  256
#define HW    3136
#define NB    32

__global__ __launch_bounds__(256) void binarize_x_nhwc(
    const float* __restrict__ x, signed char* __restrict__ xb) {
  __shared__ signed char tile[64 * 68];
  const int tid = threadIdx.x;
  const int n  = blockIdx.z;
  const int ct = blockIdx.y * 64;
  const int p0 = blockIdx.x * 64;

  {
    const int cl = tid >> 2;
    const int po = (tid & 3) * 16;
    const float* src = x + ((size_t)(n * CIN + ct + cl) * HW + p0 + po);
    signed char* trow = tile + cl * 68 + po;
#pragma unroll
    for (int j = 0; j < 4; ++j) {
      float4 v = *(const float4*)(src + j * 4);
      char4 c;
      c.x = (v.x >= 0.f) ? 1 : -1;
      c.y = (v.y >= 0.f) ? 1 : -1;
      c.z = (v.z >= 0.f) ? 1 : -1;
      c.w = (v.w >= 0.f) ? 1 : -1;
      *(char4*)(trow + j * 4) = c;
    }
  }
  __syncthreads();
  {
    const int pl = tid >> 2;
    const int co = (tid & 3) * 16;
    int wds[4];
#pragma unroll
    for (int k = 0; k < 4; ++k) {
      unsigned v = 0;
#pragma unroll
      for (int j = 0; j < 4; ++j)
        v |= (unsigned)(unsigned char)tile[(co + k * 4 + j) * 68 + pl] << (8 * j);
      wds[k] = (int)v;
    }
    int4 val = make_int4(wds[0], wds[1], wds[2], wds[3]);
    *(int4*)(xb + ((size_t)n * HW + p0 + pl) * CIN + ct + co) = val;
  }
}

__global__ __launch_bounds__(256) void pack_weights(
    const float* __restrict__ w, signed char* __restrict__ apack) {
  const int tile = blockIdx.x;
  const int t   = tile >> 6;
  const int rem = tile & 63;
  const int kb  = rem >> 4;
  const int mb  = rem & 15;
  const int tid = threadIdx.x;
  const int l   = tid >> 3;
  const int v   = tid & 7;
  const int lhalf = l >> 4;
  const int m   = mb * 16 + (l & 15);
  const int kbase = ((v & 3) << 2) | ((v >> 2) << 5);
  const int kh = t / 3, kw = t % 3;
  unsigned word = 0;
#pragma unroll
  for (int b = 0; b < 4; ++b) {
    const int k64 = kbase + lhalf * 16 + b;
    const int ci  = kb * 64 + k64;
    const float wv = w[((size_t)m * CIN + ci) * 9 + kh * 3 + kw];
    const unsigned byte = (wv >= 0.f) ? 0x01u : 0xFFu;
    word |= byte << (8 * b);
  }
  ((unsigned*)apack)[(size_t)tile * 256 + tid] = word;
}

__global__ __launch_bounds__(256) void bconv_wmma(
    const signed char* __restrict__ xb, const signed char* __restrict__ apack,
    const float* __restrict__ bias, float* __restrict__ out) {
  __shared__ alignas(16) signed char sb[2][64 * 80];

  const int tid   = threadIdx.x;
  const int lane  = tid & 31;
  const int wv    = tid >> 5;
  const int lhalf = lane >> 4;
  const int ln15  = lane & 15;
  const int n     = blockIdx.z;
  const int p0    = blockIdx.x * 64;
  const int mtile = blockIdx.y;
  const int cm    = (wv & 3);
  const int pp    = (wv >> 2);

  const int f_pix = tid >> 2;
  const int f_co  = (tid & 3) * 16;
  const int f_cd  = ((f_co == 16) ? 32 : (f_co == 32) ? 16 : f_co);
  const int p     = p0 + f_pix;
  const int h     = p / 56;
  const int w     = p % 56;
  const signed char* xnb = xb + (size_t)n * HW * CIN;

  v8i acc00 = {0, 0, 0, 0, 0, 0, 0, 0};
  v8i acc01 = {0, 0, 0, 0, 0, 0, 0, 0};
  v8i acc10 = {0, 0, 0, 0, 0, 0, 0, 0};
  v8i acc11 = {0, 0, 0, 0, 0, 0, 0, 0};

  auto fill = [&](int buf, int iter) {
    const int t  = iter >> 2;
    const int kb = iter & 3;
    const int hh = h + (t / 3) - 1;
    const int ww = w + (t % 3) - 1;
    v4i data = {0, 0, 0, 0};
    if ((unsigned)hh < 56u && (unsigned)ww < 56u) {
      data = *(const v4i*)(xnb + ((size_t)(hh * 56 + ww) * CIN) + kb * 64 + f_co);
    }
    *(v4i*)(&sb[buf][f_pix * 80 + f_cd]) = data;
  };

  fill(0, 0);
  int buf = 0;
#pragma unroll 1
  for (int iter = 0; iter < 36; ++iter) {
    __syncthreads();
    if (iter + 1 < 36) fill(buf ^ 1, iter + 1);

    const int t  = iter >> 2;
    const int kb = iter & 3;
    const int mbG0 = mtile * 8 + cm * 2;
    const size_t abase =
        (size_t)((t * 4 + kb) * 16 + mbG0) * 1024 + (size_t)lane * 32;
    if (iter + 1 < 36) {
      const int t2 = (iter + 1) >> 2, kb2 = (iter + 1) & 3;
      __builtin_prefetch(
          apack + (size_t)((t2 * 4 + kb2) * 16 + mbG0) * 1024 + lane * 32, 0, 1);
    }
    const v8i a0 = *(const v8i*)(apack + abase);
    const v8i a1 = *(const v8i*)(apack + abase + 1024);

    const signed char* srow0 = &sb[buf][(pp * 32 + ln15) * 80 + lhalf * 32];
    const signed char* srow1 = srow0 + 16 * 80;
    const v4i lo0 = *(const v4i*)(srow0);
    const v4i hi0 = *(const v4i*)(srow0 + 16);
    const v4i lo1 = *(const v4i*)(srow1);
    const v4i hi1 = *(const v4i*)(srow1 + 16);
    const v8i b0 = {lo0[0], lo0[1], lo0[2], lo0[3], hi0[0], hi0[1], hi0[2], hi0[3]};
    const v8i b1 = {lo1[0], lo1[1], lo1[2], lo1[3], hi1[0], hi1[1], hi1[2], hi1[3]};

    acc00 = __builtin_amdgcn_wmma_i32_16x16x64_iu8(true, a0, true, b0, acc00, false, false);
    acc01 = __builtin_amdgcn_wmma_i32_16x16x64_iu8(true, a0, true, b1, acc01, false, false);
    acc10 = __builtin_amdgcn_wmma_i32_16x16x64_iu8(true, a1, true, b0, acc10, false, false);
    acc11 = __builtin_amdgcn_wmma_i32_16x16x64_iu8(true, a1, true, b1, acc11, false, false);

    buf ^= 1;
  }

  const int coutBase0 = mtile * 128 + cm * 32 + lhalf * 8;
  const int pixA = p0 + pp * 32 + ln15;
#pragma unroll
  for (int r = 0; r < 8; ++r) {
    const int c0 = coutBase0 + r;
    const int c1 = c0 + 16;
    const float bv0 = bias[c0];
    const float bv1 = bias[c1];
    float* orow0 = out + ((size_t)n * COUT + c0) * HW;
    float* orow1 = out + ((size_t)n * COUT + c1) * HW;
    orow0[pixA]      = (float)acc00[r] + bv0;
    orow0[pixA + 16] = (float)acc01[r] + bv0;
    orow1[pixA]      = (float)acc10[r] + bv1;
    orow1[pixA + 16] = (float)acc11[r] + bv1;
  }
}

extern "C" void kernel_launch(void* const* d_in, const int* in_sizes, int n_in,
                              void* d_out, int out_size, void* d_ws, size_t ws_size,
                              hipStream_t stream) {
  const float* x    = (const float*)d_in[0];
  const float* wgt  = (const float*)d_in[1];
  const float* bias = (const float*)d_in[2];
  float* out        = (float*)d_out;

  signed char* xb    = (signed char*)d_ws;
  signed char* apack = (signed char*)d_ws + (size_t)NB * HW * CIN;

  dim3 blk(256, 1, 1);
  dim3 grid_bin(49, 4, 32);
  binarize_x_nhwc<<<grid_bin, blk, 0, stream>>>(x, xb);

  dim3 grid_pack(576, 1, 1);
  pack_weights<<<grid_pack, blk, 0, stream>>>(wgt, apack);

  dim3 grid_conv(49, 2, 32);
  bconv_wmma<<<grid_conv, blk, 0, stream>>>(xb, apack, bias, out);
}
